// MT_model_7172595384863
// MI455X (gfx1250) — matrix operand order repaired and checked by reading, not compiled, not run on hardware
//
#include <hip/hip_runtime.h>
#include <hip/hip_bf16.h>
#include <stdint.h>

typedef __attribute__((ext_vector_type(16))) __bf16 v16bf;
typedef __attribute__((ext_vector_type(8)))  float  v8f;

#define B_DIM 64
#define D_DIM 1024
#define L_DIM 1500
#define L_PAD 1504
#define T_DIM 78
#define C_DIM 2
#define R_REP 10
#define N_DIM (R_REP * B_DIM)    // 640
#define TC_DIM (T_DIM * C_DIM)   // 156
#define TC_PAD 160               // padded rows for det GEMM

#define SEED_Z 0xC0FFEE01u
#define SEED_W 0xDEADBEEFu

// ---------------- workspace layout (bytes) ----------------
#define WS_Z      0                                            // [640][1504] bf16
#define WS_OUTF   (WS_Z    + (size_t)N_DIM * L_PAD * 2)        // [156][640] f32
#define WS_XB     (WS_OUTF + (size_t)TC_DIM * N_DIM * 4)       // [64][1024] bf16
#define WS_X2B    (WS_XB   + (size_t)B_DIM * D_DIM * 2)        // [64][1024] bf16
#define WS_PMUT   (WS_X2B  + (size_t)B_DIM * D_DIM * 2)        // [1504][1024] bf16 (transposed)
#define WS_PVART  (WS_PMUT + (size_t)L_PAD * D_DIM * 2)        // [1504][1024] bf16 (transposed)
#define WS_WMUB   (WS_PVART+ (size_t)L_PAD * D_DIM * 2)        // [160][1504] bf16 (zero-padded)
#define WS_WSTD   (WS_WMUB + (size_t)TC_PAD * L_PAD * 2)       // [156][1504] f32  (zero-padded)

// ---------------- helpers ----------------
__device__ __forceinline__ uint32_t pcg(uint32_t v) {
    v = v * 747796405u + 2891336453u;
    uint32_t w = ((v >> ((v >> 28u) + 4u)) ^ v) * 277803737u;
    return (w >> 22u) ^ w;
}

// standard normal via PCG hash + Box-Muller (distribution-faithful stand-in for JAX threefry)
__device__ __forceinline__ float gauss(uint32_t seed, uint32_t idx) {
    uint32_t h1 = pcg(pcg(idx) ^ seed);
    uint32_t h2 = pcg(h1 + 0x9E3779B9u);
    float u1 = ((float)h1 + 1.0f) * 2.3283064365386963e-10f;  // (0,1]
    float u2 = (float)h2 * 2.3283064365386963e-10f;           // [0,1)
    float r  = sqrtf(fmaxf(-2.0f * __logf(u1), 0.0f));
    return r * __cosf(6.283185307179586f * u2);
}

// torch F.softplus(beta=1, threshold=20)
__device__ __forceinline__ float softplus20(float x) {
    return x > 20.0f ? x : log1pf(__expf(x));
}

// ---------------- prep kernels (run once, tiny) ----------------
// phi [1024 x 1500] -> transposed, bf16, zero-padded [1504 x 1024]
__global__ void k_prep_phi(const float* __restrict__ phi_mu,
                           const float* __restrict__ phi_logvar,
                           __bf16* __restrict__ pmu_T,
                           __bf16* __restrict__ pvar_T) {
    int idx = blockIdx.x * blockDim.x + threadIdx.x;
    if (idx >= L_PAD * D_DIM) return;
    int l = idx / D_DIM;
    int d = idx - l * D_DIM;
    if (l < L_DIM) {
        float m = phi_mu[d * L_DIM + l];
        float s = softplus20(phi_logvar[d * L_DIM + l]);
        pmu_T[idx]  = (__bf16)m;
        pvar_T[idx] = (__bf16)(s * s);
    } else {
        pmu_T[idx]  = (__bf16)0.0f;
        pvar_T[idx] = (__bf16)0.0f;
    }
}

// x -> bf16, x^2 -> bf16
__global__ void k_prep_x(const float* __restrict__ x,
                         __bf16* __restrict__ xb,
                         __bf16* __restrict__ x2b) {
    int idx = blockIdx.x * blockDim.x + threadIdx.x;
    if (idx >= B_DIM * D_DIM) return;
    float v = x[idx];
    xb[idx]  = (__bf16)v;
    x2b[idx] = (__bf16)(v * v);
}

// w_mu -> bf16 [160 x 1504] zero-padded; w_std = softplus(w_logvar) -> f32 [156 x 1504] zero-padded
__global__ void k_prep_w(const float* __restrict__ w_mu,
                         const float* __restrict__ w_logvar,
                         __bf16* __restrict__ wmu_b,
                         float*  __restrict__ wstd) {
    int idx = blockIdx.x * blockDim.x + threadIdx.x;
    if (idx >= TC_PAD * L_PAD) return;
    int tc = idx / L_PAD;
    int l  = idx - tc * L_PAD;
    bool ok = (tc < TC_DIM) && (l < L_DIM);
    float m = ok ? w_mu[tc * L_DIM + l] : 0.0f;
    float s = ok ? softplus20(w_logvar[tc * L_DIM + l]) : 0.0f;
    wmu_b[idx] = (__bf16)m;
    if (tc < TC_DIM) wstd[tc * L_PAD + l] = s;
}

// ---------------- stage 1: encoder (two WMMA GEMMs) + reparam sampling ----------------
// grid = (L_PAD/16 = 94, B/16 = 4), block = 32 (one wave per 16x16 tile)
__global__ void k_encoder(const __bf16* __restrict__ xb,
                          const __bf16* __restrict__ x2b,
                          const __bf16* __restrict__ pmu_T,
                          const __bf16* __restrict__ pvar_T,
                          const float* __restrict__ phi_bias_mu,
                          const float* __restrict__ phi_bias_logvar,
                          __bf16* __restrict__ z) {
    const int lane = threadIdx.x;
    const int half = lane >> 4;
    const int l15  = lane & 15;
    const int n0 = blockIdx.x * 16;    // L tile
    const int m0 = blockIdx.y * 16;    // B tile

    const int arow  = m0 + l15;                 // < 64 always
    const int akb   = half ? 16 : 0;
    const int bcol0 = n0 + l15;                 // < 1504, padded arrays: no clamp
    const int bkb   = half ? 16 : 0;

    v8f cmu = {};
    v8f cvar = {};

    for (int kk = 0; kk < D_DIM; kk += 32) {
        v16bf amu, avar, bmu, bvar;
        const __bf16* xa  = xb  + arow * D_DIM + kk + akb;
        const __bf16* xa2 = x2b + arow * D_DIM + kk + akb;
        const __bf16* pb  = pmu_T  + bcol0 * D_DIM + kk + bkb;
        const __bf16* pv  = pvar_T + bcol0 * D_DIM + kk + bkb;
#pragma unroll
        for (int j = 0; j < 8; ++j) {
            amu[j]      = xa[j];
            amu[8 + j]  = xa[8 + j];
            avar[j]     = xa2[j];
            avar[8 + j] = xa2[8 + j];
        }
#pragma unroll
        for (int j = 0; j < 16; ++j) {
            bmu[j]  = pb[j];
            bvar[j] = pv[j];
        }
        cmu  = __builtin_amdgcn_wmma_f32_16x16x32_bf16(false, amu,  false, bmu,
                                                       (short)0, cmu,  false, false);
        cvar = __builtin_amdgcn_wmma_f32_16x16x32_bf16(false, avar, false, bvar,
                                                       (short)0, cvar, false, false);
    }

    const int  bcol  = bcol0 < L_DIM ? bcol0 : (L_DIM - 1);  // clamp only bias loads
    const bool colok = (bcol0 < L_DIM);
    const float bias_mu  = phi_bias_mu[bcol];
    const float pbs      = softplus20(phi_bias_logvar[bcol]);
    const float bias_var = pbs * pbs;

#pragma unroll
    for (int i = 0; i < 8; ++i) {
        int   m    = m0 + i + (half ? 8 : 0);
        float zmu  = cmu[i] + bias_mu;
        float zstd = sqrtf(fmaxf(cvar[i] + bias_var, 0.0f));
#pragma unroll
        for (int r = 0; r < R_REP; ++r) {
            int n = r * B_DIM + m;
            float eps = gauss(SEED_Z, (uint32_t)(n * L_PAD + bcol0));
            float zv  = zmu + zstd * eps;
            z[n * L_PAD + bcol0] = colok ? (__bf16)zv : (__bf16)0.0f;  // pad cols -> 0
        }
    }
}

// ---------------- stage 2: deterministic term, WMMA GEMM [160 x 1504] * [1504 x 640] ----------------
// grid = (N/16 = 40, TC_PAD/16 = 10), block = 32
__global__ void k_det_gemm(const __bf16* __restrict__ wmu_b,
                           const __bf16* __restrict__ z,
                           float* __restrict__ out_full) {
    const int lane = threadIdx.x;
    const int half = lane >> 4;
    const int l15  = lane & 15;
    const int n0 = blockIdx.x * 16;
    const int m0 = blockIdx.y * 16;

    const int arow = m0 + l15;                  // < 160, padded: no clamp
    const int akb  = half ? 16 : 0;
    const int bn   = n0 + l15;                  // < 640 always
    const int bkb  = half ? 16 : 0;

    v8f acc = {};
    for (int kk = 0; kk < L_PAD; kk += 32) {
        v16bf a, b;
        const __bf16* ap = wmu_b + arow * L_PAD + kk + akb;
        const __bf16* bp = z     + bn   * L_PAD + kk + bkb;
#pragma unroll
        for (int j = 0; j < 8; ++j) {
            a[j]     = ap[j];
            a[8 + j] = ap[8 + j];
        }
#pragma unroll
        for (int j = 0; j < 16; ++j) {
            b[j] = bp[j];
        }
        acc = __builtin_amdgcn_wmma_f32_16x16x32_bf16(false, a, false, b,
                                                      (short)0, acc, false, false);
    }
#pragma unroll
    for (int i = 0; i < 8; ++i) {
        int m = m0 + i + (half ? 8 : 0);
        if (m < TC_DIM) out_full[m * N_DIM + bn] = acc[i];
    }
}

// ---------------- stage 3: stochastic term (dominant cost, pure VALU) ----------------
// one wave per (tc, n): 156*640 = 99840 waves; block = 256 (8 waves)
__global__ void k_noise(const float* __restrict__ wstd,
                        const __bf16* __restrict__ z,
                        float* __restrict__ out_full) {
    const int gid  = blockIdx.x * blockDim.x + threadIdx.x;
    const int wave = gid >> 5;
    const int lane = gid & 31;
    if (wave >= TC_DIM * N_DIM) return;

    const int tc = wave / N_DIM;
    const int n  = wave - tc * N_DIM;
    const int t  = tc >> 1;
    const int c  = tc & 1;
    // eps_w flat index base: ((t*N + n)*C + c)*L,  T*N*C*L = 149.76M < 2^32
    const uint32_t base = ((uint32_t)(t * N_DIM + n) * (uint32_t)C_DIM + (uint32_t)c)
                          * (uint32_t)L_DIM;

    const float*  wp = wstd + tc * L_PAD;
    const __bf16* zp = z    + n  * L_PAD;

    float acc = 0.0f;
    for (int l = lane; l < L_DIM; l += 32) {
        float ws  = wp[l];
        float zv  = (float)zp[l];
        float eps = gauss(SEED_W, base + (uint32_t)l);
        acc = fmaf(ws * eps, zv, acc);
    }
#pragma unroll
    for (int off = 16; off > 0; off >>= 1) acc += __shfl_xor(acc, off, 32);
    if (lane == 0) out_full[tc * N_DIM + n] += acc;
}

// ---------------- stage 4: mean over R + transpose to [B, T, C] ----------------
__global__ void k_reduce(const float* __restrict__ out_full,
                         float* __restrict__ out) {
    const int idx = blockIdx.x * blockDim.x + threadIdx.x;
    if (idx >= B_DIM * TC_DIM) return;
    const int b  = idx / TC_DIM;
    const int tc = idx - b * TC_DIM;
    float s = 0.0f;
#pragma unroll
    for (int r = 0; r < R_REP; ++r) s += out_full[tc * N_DIM + r * B_DIM + b];
    out[idx] = s * (1.0f / (float)R_REP);
}

extern "C" void kernel_launch(void* const* d_in, const int* in_sizes, int n_in,
                              void* d_out, int out_size, void* d_ws, size_t ws_size,
                              hipStream_t stream) {
    const float* x               = (const float*)d_in[0];
    const float* phi_mu          = (const float*)d_in[1];
    const float* phi_logvar      = (const float*)d_in[2];
    const float* phi_bias_mu     = (const float*)d_in[3];
    const float* phi_bias_logvar = (const float*)d_in[4];
    const float* w_mu            = (const float*)d_in[5];
    const float* w_logvar        = (const float*)d_in[6];
    float* out = (float*)d_out;

    char* ws = (char*)d_ws;
    __bf16* z        = (__bf16*)(ws + WS_Z);
    float*  out_full = (float*) (ws + WS_OUTF);
    __bf16* xb       = (__bf16*)(ws + WS_XB);
    __bf16* x2b      = (__bf16*)(ws + WS_X2B);
    __bf16* pmu_T    = (__bf16*)(ws + WS_PMUT);
    __bf16* pvar_T   = (__bf16*)(ws + WS_PVART);
    __bf16* wmu_b    = (__bf16*)(ws + WS_WMUB);
    float*  wstd     = (float*) (ws + WS_WSTD);

    // ---- prep (once per call, tiny) ----
    {
        int n = L_PAD * D_DIM;
        k_prep_phi<<<(n + 255) / 256, 256, 0, stream>>>(phi_mu, phi_logvar, pmu_T, pvar_T);
    }
    {
        int n = B_DIM * D_DIM;
        k_prep_x<<<(n + 255) / 256, 256, 0, stream>>>(x, xb, x2b);
    }
    {
        int n = TC_PAD * L_PAD;
        k_prep_w<<<(n + 255) / 256, 256, 0, stream>>>(w_mu, w_logvar, wmu_b, wstd);
    }

    // ---- pipeline ----
    dim3 g1(L_PAD / 16, B_DIM / 16);            // (94, 4)
    k_encoder<<<g1, 32, 0, stream>>>(xb, x2b, pmu_T, pvar_T,
                                     phi_bias_mu, phi_bias_logvar, z);

    dim3 g2(N_DIM / 16, TC_PAD / 16);           // (40, 10)
    k_det_gemm<<<g2, 32, 0, stream>>>(wmu_b, z, out_full);

    const int nwaves = TC_DIM * N_DIM;          // 99840
    k_noise<<<(nwaves * 32 + 255) / 256, 256, 0, stream>>>(wstd, z, out_full);

    const int nout = B_DIM * TC_DIM;            // 9984
    k_reduce<<<(nout + 255) / 256, 256, 0, stream>>>(out_full, out);
}
